// GrayScaleDifferenceEnhancement_6648609374643
// MI455X (gfx1250) — matrix operand order repaired and checked by reading, not compiled, not run on hardware
//
#include <hip/hip_runtime.h>
#include <hip/hip_bf16.h>
#include <math.h>

// ---------------------------------------------------------------------------
// Types for WMMA fragments (CDNA5 gfx1250, wave32)
// ---------------------------------------------------------------------------
typedef _Float16 half16 __attribute__((ext_vector_type(16)));
typedef _Float16 half8  __attribute__((ext_vector_type(8)));
typedef float    f32x8  __attribute__((ext_vector_type(8)));

#define BN_SCALE 0.9999950000374996f  // 1/sqrt(1+1e-5)

__device__ __forceinline__ half16 h16_combine(const half8 lo, const half8 hi) {
  half16 a;
#pragma unroll
  for (int e = 0; e < 8; ++e) { a[e] = lo[e]; a[e + 8] = hi[e]; }
  return a;
}

// Async global -> LDS 16-byte copy (gfx1250, tracked by ASYNCcnt)
__device__ __forceinline__ void async_copy_b128(uint32_t lds_byte_off, const void* gptr) {
  asm volatile("global_load_async_to_lds_b128 %0, %1, off"
               :: "v"(lds_byte_off), "v"(gptr)
               : "memory");
}

__device__ __forceinline__ void wait_async0() {
#if __has_builtin(__builtin_amdgcn_s_wait_asynccnt)
  __builtin_amdgcn_s_wait_asynccnt(0);
#else
  asm volatile("s_wait_asynccnt 0x0" ::: "memory");
#endif
}

// ---------------------------------------------------------------------------
// Implicit-GEMM 3x3 conv, stride 1, pad 1 (via halo), BN scale + ReLU, f16 I/O
// Activations: NHWC with 1-px halo:  [8][H+2][W+2][CIN]
// Weights packed: [tap=9][CO][CI] f16; FIRST layer packs taps-as-K: [CO][32]
// Block: 256 threads = 8 waves; 4 waves tile M (64 co), 2 waves tile N;
// each wave computes 16 co x 64 px via 4 accumulators of v_wmma_f32_16x16x32_f16.
// Per K-chunk the [9][64][32] f16 weight tile (36 KB) is staged into LDS with
// global_load_async_to_lds_b128; A-fragments are then ds_load'ed from LDS.
// ---------------------------------------------------------------------------
template <int CI, int CO, bool FIRST>
__global__ __launch_bounds__(256) void conv3x3_bnrelu(
    const _Float16* __restrict__ in, const _Float16* __restrict__ wpk,
    _Float16* __restrict__ out, int H, int W) {
  constexpr int CIN = FIRST ? 1 : CI;
  __shared__ _Float16 wlds[FIRST ? 16 : 9 * 64 * 32];  // 36 KB weight stage
  const int n     = blockIdx.z;
  const int cob   = blockIdx.y * 64;
  const int wave  = threadIdx.x >> 5;
  const int lane  = threadIdx.x & 31;
  const int mwave = wave & 3;   // 4 co sub-tiles of 16
  const int nwave = wave >> 2;  // 2 pixel halves of 64
  const int laneM = lane & 15;
  const int hi    = lane >> 4;  // 0 or 1 (wave32 half)
  const int HP = H + 2, WP = W + 2;
  const _Float16* inb = in + (size_t)n * HP * WP * CIN;
  const int co = cob + mwave * 16 + laneM;  // A-matrix row for this lane

  int py[4], px[4];
  const int p0 = blockIdx.x * 128 + nwave * 64;
#pragma unroll
  for (int j = 0; j < 4; ++j) {
    int p = p0 + j * 16;          // W % 16 == 0 -> group stays in one row
    py[j] = p / W;
    px[j] = (p % W) + laneM;
  }

  f32x8 acc[4];
#pragma unroll
  for (int j = 0; j < 4; ++j)
#pragma unroll
    for (int e = 0; e < 8; ++e) acc[j][e] = 0.f;

  if constexpr (FIRST) {
    // Ci==1: taps ARE the K dimension (K=9 zero-padded to 32); one WMMA/group.
    const _Float16* wp = wpk + (size_t)co * 32 + hi * 16;  // [co][k]
    half16 a = h16_combine(*(const half8*)wp, *(const half8*)(wp + 8));
#pragma unroll
    for (int j = 0; j < 4; ++j) {
      half16 b;
#pragma unroll
      for (int e = 0; e < 16; ++e) b[e] = (_Float16)0;
      if (hi == 0) {  // K=0..15 half holds the 9 taps
#pragma unroll
        for (int e = 0; e < 9; ++e)
          b[e] = inb[(size_t)(py[j] + e / 3) * WP + (px[j] + e % 3)];
      }
      acc[j] = __builtin_amdgcn_wmma_f32_16x16x32_f16(
          false, a, false, b, (short)0, acc[j], false, false);
    }
  } else {
    const uint32_t lds_base = (uint32_t)(uintptr_t)(&wlds[0]);
    const int KC = CI / 32;
    for (int kc = 0; kc < KC; ++kc) {
      __syncthreads();  // previous chunk's readers done before re-staging
      // Stage [9][64][32] weight chunk: 2304 x 16B chunks == 9 per thread.
#pragma unroll
      for (int it = 0; it < 9; ++it) {
        int q = threadIdx.x + it * 256;
        int part = q & 3;           // 4 x 8 halves per (tap,co) row
        int row  = q >> 2;          // row = tap*64 + co_local
        int tap  = row >> 6;
        int col  = row & 63;
        const _Float16* gsrc =
            wpk + ((size_t)tap * CO + cob + col) * CI + kc * 32 + part * 8;
        async_copy_b128(lds_base + (uint32_t)q * 16, (const void*)gsrc);
      }
      wait_async0();
      __syncthreads();

      if (kc + 1 < KC)  // hint next K-chunk of activations
        __builtin_prefetch(inb + ((size_t)py[0] * WP + px[0]) * CI + (kc + 1) * 32, 0, 0);

#pragma unroll
      for (int r = 0; r < 3; ++r) {
#pragma unroll
        for (int s = 0; s < 3; ++s) {
          const int tap = r * 3 + s;
          // A fragment from LDS: lane<16 -> K = 0..15 ; lane>=16 -> K = 16..31
          const _Float16* wp = wlds + ((tap * 64 + mwave * 16 + laneM) * 32) + hi * 16;
          half16 a = h16_combine(*(const half8*)wp, *(const half8*)(wp + 8));
#pragma unroll
          for (int j = 0; j < 4; ++j) {
            // B fragment: 16 contiguous halves along ci (lane half picks K base)
            half16 b = *(const half16*)(inb +
                ((size_t)(py[j] + r) * WP + (px[j] + s)) * CI + kc * 32 + hi * 16);
            acc[j] = __builtin_amdgcn_wmma_f32_16x16x32_f16(
                false, a, false, b, (short)0, acc[j], false, false);
          }
        }
      }
    }
  }

  // Epilogue: BN scale + ReLU, pack 8 consecutive co per lane (C/D layout)
#pragma unroll
  for (int j = 0; j < 4; ++j) {
    half8 o;
#pragma unroll
    for (int rr = 0; rr < 8; ++rr) {
      float v = acc[j][rr] * BN_SCALE;
      o[rr] = (_Float16)fmaxf(v, 0.f);
    }
    size_t off = (((size_t)n * HP + (py[j] + 1)) * WP + (px[j] + 1)) * CO +
                 cob + mwave * 16 + hi * 8;
    *(half8*)(out + off) = o;
  }
}

// ---------------------------------------------------------------------------
// Weight repack: OIHW f32 -> [tap][CO][CI] f16
// ---------------------------------------------------------------------------
__global__ void repack_w(const float* __restrict__ w, _Float16* __restrict__ out,
                         int CO, int CI, int CIP) {
  int idx = blockIdx.x * 256 + threadIdx.x;
  int total = 9 * CO * CIP;
  if (idx >= total) return;
  int ci = idx % CIP;
  int t  = idx / CIP;
  int co = t % CO;
  int tap = t / CO;
  int r = tap / 3, s = tap % 3;
  float v = (ci < CI) ? w[((co * CI + ci) * 3 + r) * 3 + s] : 0.f;
  out[idx] = (_Float16)v;
}

// First layer (Ci=1): taps-as-K packing  out[co][k] = w[co][tap=k], k<9
__global__ void repack_e00_k(const float* __restrict__ w, _Float16* __restrict__ out) {
  int idx = blockIdx.x * 256 + threadIdx.x;
  if (idx >= 64 * 32) return;
  int k = idx & 31, co = idx >> 5;
  out[idx] = (_Float16)((k < 9) ? w[co * 9 + k] : 0.f);
}

// ---------------------------------------------------------------------------
// Zero halo ring of an NHWC-with-halo buffer (8 images)
// ---------------------------------------------------------------------------
template <typename T>
__global__ void zero_halo_k(T* buf, int H, int W, int C) {
  size_t rowelems = (size_t)(W + 2) * C;
  size_t per_n = 2 * rowelems + (size_t)2 * H * C;
  size_t total = 8 * per_n;
  size_t idx = (size_t)blockIdx.x * 256 + threadIdx.x;
  if (idx >= total) return;
  int n = (int)(idx / per_n);
  size_t r = idx % per_n;
  T* b = buf + (size_t)n * (H + 2) * (W + 2) * C;
  if (r < rowelems) { b[r] = (T)0; return; }
  r -= rowelems;
  if (r < rowelems) { b[(size_t)(H + 1) * (W + 2) * C + r] = (T)0; return; }
  r -= rowelems;
  size_t c = r % C;
  size_t t = r / C;
  int side = (int)(t & 1);
  int y = (int)(t >> 1);
  size_t x = side ? (size_t)(W + 1) : 0;
  b[((size_t)(y + 1) * (W + 2) + x) * C + c] = (T)0;
}

// ---------------------------------------------------------------------------
// Gray -> complexity -> ks  (one block per image)
// ---------------------------------------------------------------------------
__global__ __launch_bounds__(256) void complexity_ks_k(const float* __restrict__ x,
                                                       float* __restrict__ comp_out,
                                                       int* __restrict__ ks) {
  int n = blockIdx.x;
  const float* xb = x + (size_t)n * 3 * 65536;
  float s = 0.f;
  for (int p = threadIdx.x; p < 256 * 255; p += 256) {
    int h = p / 255, w = p % 255;
    int i0 = h * 256 + w;
    float g0 = floorf((0.299f * xb[i0] + 0.587f * xb[65536 + i0] + 0.114f * xb[131072 + i0]) * 255.f);
    float g1 = floorf((0.299f * xb[i0 + 1] + 0.587f * xb[65536 + i0 + 1] + 0.114f * xb[131072 + i0 + 1]) * 255.f);
    float d = g1 - g0;
    s += d * d;
  }
  __shared__ float red[256];
  red[threadIdx.x] = s;
  __syncthreads();
  for (int o = 128; o > 0; o >>= 1) {
    if ((int)threadIdx.x < o) red[threadIdx.x] += red[threadIdx.x + o];
    __syncthreads();
  }
  if (threadIdx.x == 0) {
    float c = red[0] / (256.f * 255.f);
    comp_out[n] = c;
    int base = (c < 50.f) ? 30 : ((c < 150.f) ? 80 : 150);
    int inc  = (c < 50.f) ? 10 : ((c < 150.f) ? 15 : 25);
    ks[n * 3 + 0] = base - inc;
    ks[n * 3 + 1] = base;
    ks[n * 3 + 2] = base + inc;
  }
}

// pix = mean over channels of floor(x*255)
__global__ void pix_k(const float* __restrict__ x, float* __restrict__ pix) {
  size_t i = (size_t)blockIdx.x * 256 + threadIdx.x;
  if (i >= (size_t)8 * 65536) return;
  size_t n = i >> 16, p = i & 65535;
  const float* xb = x + n * 3 * 65536;
  pix[i] = (floorf(xb[p] * 255.f) + floorf(xb[65536 + p] * 255.f) +
            floorf(xb[131072 + p] * 255.f)) * (1.f / 3.f);
}

// ---------------------------------------------------------------------------
// grid_saliency: one block per (image n, branch i); LDS segment reduction.
// Segment counts are analytic (band sizes), so only sums need atomics.
// ---------------------------------------------------------------------------
__global__ __launch_bounds__(256) void mss_k(const float* __restrict__ pix,
                                             const int* __restrict__ ks,
                                             float* __restrict__ mss_out,
                                             _Float16* __restrict__ bin0,
                                             _Float16* __restrict__ bin1,
                                             _Float16* __restrict__ bin2) {
  int n = blockIdx.x, i = blockIdx.y;
  int k = ks[n * 3 + i];
  int g = (int)floorf(sqrtf((float)k));
  if (g < 1) g = 1;
  int ch = (256 + g - 1) / g;
  int cw = ch;
  __shared__ float sums[196];
  for (int t = threadIdx.x; t < 196; t += 256) sums[t] = 0.f;
  __syncthreads();
  const float* pb = pix + (size_t)n * 65536;
  for (int p = threadIdx.x; p < 65536; p += 256) {
    int h = p >> 8, w = p & 255;
    int seg = (h / ch) * g + (w / cw);
    atomicAdd(&sums[seg], pb[p]);
  }
  __syncthreads();
  for (int t = threadIdx.x; t < 196; t += 256) {
    int gy = t / g, gx = t % g;
    float m = 0.f;
    if (gy < g) {
      int rows = 256 - gy * ch; if (rows > ch) rows = ch; if (rows < 0) rows = 0;
      int cols = 256 - gx * cw; if (cols > cw) cols = cw; if (cols < 0) cols = 0;
      m = sums[t] / fmaxf((float)rows * (float)cols, 1.f);
    }
    sums[t] = m;
  }
  __syncthreads();
  _Float16* bin = (i == 0) ? bin0 : ((i == 1) ? bin1 : bin2);
  _Float16* bb = bin + (size_t)n * 258 * 258;
  float* mo = mss_out + ((size_t)n * 3 + i) * 65536;
  for (int p = threadIdx.x; p < 65536; p += 256) {
    int h = p >> 8, w = p & 255;
    float v = sums[(h / ch) * g + (w / cw)] * (1.f / 255.f);
    mo[p] = v;
    bb[(size_t)(h + 1) * 258 + (w + 1)] = (_Float16)v;
  }
}

// 2x2 max pool, NHWC-halo f16 in/out
__global__ void maxpool2_k(const _Float16* __restrict__ in, _Float16* __restrict__ out,
                           int H, int W, int C) {
  int OH = H / 2, OW = W / 2;
  size_t total = (size_t)8 * OH * OW * C;
  size_t idx = (size_t)blockIdx.x * 256 + threadIdx.x;
  if (idx >= total) return;
  int c = (int)(idx % C);
  size_t t = idx / C;
  int ox = (int)(t % OW); t /= OW;
  int oy = (int)(t % OH);
  int n  = (int)(t / OH);
  const _Float16* ib = in + (size_t)n * (H + 2) * (W + 2) * C;
  const int WP = W + 2;
  float a = (float)ib[(((size_t)(2 * oy + 1)) * WP + (2 * ox + 1)) * C + c];
  float b = (float)ib[(((size_t)(2 * oy + 1)) * WP + (2 * ox + 2)) * C + c];
  float d = (float)ib[(((size_t)(2 * oy + 2)) * WP + (2 * ox + 1)) * C + c];
  float e = (float)ib[(((size_t)(2 * oy + 2)) * WP + (2 * ox + 2)) * C + c];
  float m = fmaxf(fmaxf(a, b), fmaxf(d, e));
  out[(((size_t)n * (OH + 2) + (oy + 1)) * (OW + 2) + (ox + 1)) * C + c] = (_Float16)m;
}

// bilinear x2 upsample (align-corners weights per reference) + residual add
__global__ void up2add_k(const _Float16* __restrict__ in, const _Float16* __restrict__ res,
                         _Float16* __restrict__ out, int h, int C, int Cout, int coff) {
  const int oh = 2 * h, ow = 2 * h;
  size_t total = (size_t)8 * oh * ow * C;
  size_t idx = (size_t)blockIdx.x * 256 + threadIdx.x;
  if (idx >= total) return;
  int c = (int)(idx % C);
  size_t t = idx / C;
  int ox = (int)(t % ow); t /= ow;
  int oy = (int)(t % oh);
  int n  = (int)(t / oh);
  const float sc = (float)(h - 1) / (float)(oh - 1);
  float fy = oy * sc;
  int y0 = (int)floorf(fy);
  int y1 = y0 + 1; if (y1 > h - 1) y1 = h - 1;
  float wy = fy - (float)y0;
  float fx = ox * sc;
  int x0 = (int)floorf(fx);
  int x1 = x0 + 1; if (x1 > h - 1) x1 = h - 1;
  float wx = fx - (float)x0;
  const _Float16* ib = in + (size_t)n * (h + 2) * (h + 2) * C;
  const int WP = h + 2;
  float v00 = (float)ib[(((size_t)(y0 + 1)) * WP + (x0 + 1)) * C + c];
  float v10 = (float)ib[(((size_t)(y1 + 1)) * WP + (x0 + 1)) * C + c];
  float v01 = (float)ib[(((size_t)(y0 + 1)) * WP + (x1 + 1)) * C + c];
  float v11 = (float)ib[(((size_t)(y1 + 1)) * WP + (x1 + 1)) * C + c];
  float r0 = v00 * (1.f - wy) + v10 * wy;
  float r1 = v01 * (1.f - wy) + v11 * wy;
  float v = r0 * (1.f - wx) + r1 * wx;
  v += (float)res[(((size_t)n * (oh + 2) + (oy + 1)) * (ow + 2) + (ox + 1)) * C + c];
  out[(((size_t)n * (oh + 2) + (oy + 1)) * (ow + 2) + (ox + 1)) * Cout + coff + c] = (_Float16)v;
}

// 1x1 conv (64 -> 1) + sigmoid; writes f32 into haloed (8,130,130) buffer
__global__ void conv1x1_sig_k(const _Float16* __restrict__ in, const float* __restrict__ w,
                              float* __restrict__ s) {
  size_t idx = (size_t)blockIdx.x * 256 + threadIdx.x;
  if (idx >= (size_t)8 * 128 * 128) return;
  int x = (int)(idx % 128);
  int y = (int)((idx / 128) % 128);
  int n = (int)(idx / 16384);
  const _Float16* ib = in + (((size_t)n * 130 + (y + 1)) * 130 + (x + 1)) * 64;
  float acc = 0.f;
#pragma unroll
  for (int c = 0; c < 64; ++c) acc += (float)ib[c] * w[c];
  s[((size_t)n * 130 + (y + 1)) * 130 + (x + 1)] = 1.f / (1.f + expf(-acc));
}

// final 3x3 conv (1 -> 1, pad 1) on haloed f32 buffer -> d_out enhanced
__global__ void conv_d_k(const float* __restrict__ s, const float* __restrict__ wd,
                         float* __restrict__ outE) {
  size_t idx = (size_t)blockIdx.x * 256 + threadIdx.x;
  if (idx >= (size_t)8 * 128 * 128) return;
  int x = (int)(idx % 128);
  int y = (int)((idx / 128) % 128);
  int n = (int)(idx / 16384);
  const float* sb = s + (size_t)n * 130 * 130;
  float acc = 0.f;
#pragma unroll
  for (int r = 0; r < 3; ++r)
#pragma unroll
    for (int c = 0; c < 3; ++c)
      acc += sb[(size_t)(y + r) * 130 + (x + c)] * wd[r * 3 + c];
  outE[(size_t)n * 16384 + y * 128 + x] = acc;
}

// ---------------------------------------------------------------------------
// Host orchestration
// ---------------------------------------------------------------------------
static inline size_t cdivz(size_t a, size_t b) { return (a + b - 1) / b; }

extern "C" void kernel_launch(void* const* d_in, const int* in_sizes, int n_in,
                              void* d_out, int out_size, void* d_ws, size_t ws_size,
                              hipStream_t stream) {
  (void)in_sizes; (void)n_in; (void)out_size; (void)ws_size;
  const float* x     = (const float*)d_in[0];
  const float* w_e00 = (const float*)d_in[1];
  const float* w_e01 = (const float*)d_in[2];
  const float* w_e10 = (const float*)d_in[3];
  const float* w_e11 = (const float*)d_in[4];
  const float* w_e20 = (const float*)d_in[5];
  const float* w_e21 = (const float*)d_in[6];
  const float* w_u0  = (const float*)d_in[7];
  const float* w_u1  = (const float*)d_in[8];
  const float* w_f0  = (const float*)d_in[9];
  const float* w_f1  = (const float*)d_in[10];
  const float* w_d   = (const float*)d_in[11];

  float* out_enh  = (float*)d_out;                       // 8*128*128
  float* out_mss  = (float*)d_out + 131072;              // 8*3*256*256
  float* out_comp = (float*)d_out + 131072 + 1572864;    // 8

  // Workspace bump allocator (deterministic)
  char* ws = (char*)d_ws;
  size_t off = 0;
  auto alloc = [&](size_t bytes) -> char* {
    char* p = ws + off;
    off = (off + bytes + 255) & ~(size_t)255;
    return p;
  };
  int*       ks    = (int*)alloc(24 * sizeof(int));
  float*     pix   = (float*)alloc((size_t)8 * 65536 * 4);
  _Float16*  pe00  = (_Float16*)alloc((size_t)64 * 32 * 2);
  _Float16*  pe01  = (_Float16*)alloc((size_t)9 * 64 * 64 * 2);
  _Float16*  pe10  = (_Float16*)alloc((size_t)9 * 128 * 64 * 2);
  _Float16*  pe11  = (_Float16*)alloc((size_t)9 * 128 * 128 * 2);
  _Float16*  pe20  = (_Float16*)alloc((size_t)9 * 256 * 128 * 2);
  _Float16*  pe21  = (_Float16*)alloc((size_t)9 * 256 * 256 * 2);
  _Float16*  pu0   = (_Float16*)alloc((size_t)9 * 128 * 256 * 2);
  _Float16*  pu1   = (_Float16*)alloc((size_t)9 * 64 * 128 * 2);
  _Float16*  pf0   = (_Float16*)alloc((size_t)9 * 64 * 192 * 2);
  _Float16*  bin[3];
  for (int i = 0; i < 3; ++i) bin[i] = (_Float16*)alloc((size_t)8 * 258 * 258 * 2);
  _Float16*  T0    = (_Float16*)alloc((size_t)8 * 258 * 258 * 64 * 2);
  _Float16*  T1    = (_Float16*)alloc((size_t)8 * 258 * 258 * 64 * 2);
  _Float16*  P1    = (_Float16*)alloc((size_t)8 * 130 * 130 * 64 * 2);
  _Float16*  P2    = (_Float16*)alloc((size_t)8 * 66 * 66 * 128 * 2);
  _Float16*  P3    = (_Float16*)alloc((size_t)8 * 34 * 34 * 256 * 2);
  _Float16*  U0O   = (_Float16*)alloc((size_t)8 * 34 * 34 * 128 * 2);
  _Float16*  D1    = (_Float16*)alloc((size_t)8 * 66 * 66 * 128 * 2);
  _Float16*  U1O   = (_Float16*)alloc((size_t)8 * 66 * 66 * 64 * 2);
  _Float16*  FUSED = (_Float16*)alloc((size_t)8 * 130 * 130 * 192 * 2);
  _Float16*  FCONV = (_Float16*)alloc((size_t)8 * 130 * 130 * 64 * 2);
  float*     SBUF  = (float*)alloc((size_t)8 * 130 * 130 * 4);

  auto zero_halo_h = [&](_Float16* buf, int H, int W, int C) {
    size_t total = (size_t)8 * ((size_t)C * (2 * (W + 2) + 2 * H));
    zero_halo_k<_Float16><<<(unsigned)cdivz(total, 256), 256, 0, stream>>>(buf, H, W, C);
  };
  auto zero_halo_f = [&](float* buf, int H, int W, int C) {
    size_t total = (size_t)8 * ((size_t)C * (2 * (W + 2) + 2 * H));
    zero_halo_k<float><<<(unsigned)cdivz(total, 256), 256, 0, stream>>>(buf, H, W, C);
  };
  auto repack = [&](const float* w, _Float16* outp, int CO, int CI, int CIP) {
    size_t total = (size_t)9 * CO * CIP;
    repack_w<<<(unsigned)cdivz(total, 256), 256, 0, stream>>>(w, outp, CO, CI, CIP);
  };

  // Stage 0: stats + saliency inputs
  complexity_ks_k<<<8, 256, 0, stream>>>(x, out_comp, ks);
  pix_k<<<(unsigned)cdivz((size_t)8 * 65536, 256), 256, 0, stream>>>(x, pix);

  // Weight repack (f32 OIHW -> f16 [tap][co][ci]; first layer taps-as-K)
  repack_e00_k<<<(unsigned)cdivz((size_t)64 * 32, 256), 256, 0, stream>>>(w_e00, pe00);
  repack(w_e01, pe01, 64, 64, 64);
  repack(w_e10, pe10, 128, 64, 64);
  repack(w_e11, pe11, 128, 128, 128);
  repack(w_e20, pe20, 256, 128, 128);
  repack(w_e21, pe21, 256, 256, 256);
  repack(w_u0,  pu0,  128, 256, 256);
  repack(w_u1,  pu1,  64, 128, 128);
  repack(w_f0,  pf0,  64, 192, 192);

  // Branch inputs (saliency maps)
  for (int i = 0; i < 3; ++i) zero_halo_h(bin[i], 256, 256, 1);
  mss_k<<<dim3(8, 3), 256, 0, stream>>>(pix, ks, out_mss, bin[0], bin[1], bin[2]);

  zero_halo_h(FUSED, 128, 128, 192);

  auto pool = [&](const _Float16* in, _Float16* outp, int H, int W, int C) {
    size_t total = (size_t)8 * (H / 2) * (W / 2) * C;
    maxpool2_k<<<(unsigned)cdivz(total, 256), 256, 0, stream>>>(in, outp, H, W, C);
  };

  // 3 encoder/decoder branches (sequential buffer reuse)
  for (int i = 0; i < 3; ++i) {
    zero_halo_h(T0, 256, 256, 64);
    conv3x3_bnrelu<32, 64, true><<<dim3(65536 / 128, 1, 8), 256, 0, stream>>>(bin[i], pe00, T0, 256, 256);
    conv3x3_bnrelu<64, 64, false><<<dim3(65536 / 128, 1, 8), 256, 0, stream>>>(T0, pe01, T1, 256, 256);
    zero_halo_h(P1, 128, 128, 64);
    pool(T1, P1, 256, 256, 64);

    zero_halo_h(T0, 128, 128, 128);
    conv3x3_bnrelu<64, 128, false><<<dim3(16384 / 128, 2, 8), 256, 0, stream>>>(P1, pe10, T0, 128, 128);
    conv3x3_bnrelu<128, 128, false><<<dim3(16384 / 128, 2, 8), 256, 0, stream>>>(T0, pe11, T1, 128, 128);
    zero_halo_h(P2, 64, 64, 128);
    pool(T1, P2, 128, 128, 128);

    zero_halo_h(T0, 64, 64, 256);
    conv3x3_bnrelu<128, 256, false><<<dim3(4096 / 128, 4, 8), 256, 0, stream>>>(P2, pe20, T0, 64, 64);
    conv3x3_bnrelu<256, 256, false><<<dim3(4096 / 128, 4, 8), 256, 0, stream>>>(T0, pe21, T1, 64, 64);
    zero_halo_h(P3, 32, 32, 256);
    pool(T1, P3, 64, 64, 256);

    zero_halo_h(U0O, 32, 32, 128);
    conv3x3_bnrelu<256, 128, false><<<dim3(1024 / 128, 2, 8), 256, 0, stream>>>(P3, pu0, U0O, 32, 32);
    zero_halo_h(D1, 64, 64, 128);
    {
      size_t total = (size_t)8 * 64 * 64 * 128;
      up2add_k<<<(unsigned)cdivz(total, 256), 256, 0, stream>>>(U0O, P2, D1, 32, 128, 128, 0);
    }
    conv3x3_bnrelu<128, 64, false><<<dim3(4096 / 128, 1, 8), 256, 0, stream>>>(D1, pu1, U1O, 64, 64);
    {
      size_t total = (size_t)8 * 128 * 128 * 64;
      up2add_k<<<(unsigned)cdivz(total, 256), 256, 0, stream>>>(U1O, P1, FUSED, 64, 64, 192, 64 * i);
    }
  }

  // Fuse head
  conv3x3_bnrelu<192, 64, false><<<dim3(16384 / 128, 1, 8), 256, 0, stream>>>(FUSED, pf0, FCONV, 128, 128);
  zero_halo_f(SBUF, 128, 128, 1);
  conv1x1_sig_k<<<(unsigned)cdivz((size_t)8 * 16384, 256), 256, 0, stream>>>(FCONV, w_f1, SBUF);
  conv_d_k<<<(unsigned)cdivz((size_t)8 * 16384, 256), 256, 0, stream>>>(SBUF, w_d, out_enh);
}
